// Decoder_34033320853724
// MI455X (gfx1250) — matrix operand order repaired and checked by reading, not compiled, not run on hardware
//
#include <hip/hip_runtime.h>
#include <hip/hip_bf16.h>

typedef __attribute__((ext_vector_type(16))) __bf16 bf16x16;
typedef __attribute__((ext_vector_type(8)))  __bf16 bf16x8;
typedef __attribute__((ext_vector_type(8)))  float  f32x8;

#define B_  64
#define T_  50
#define E_  300
#define EP_ 320     // E padded to multiple of 32
#define H_  1024
#define V_  30000
#define DINIT_ 824
#define DATT_  200
#define LDS_COL_STRIDE 2064   // 2048B + 16B pad: bank-conflict-free b128 reads

static __device__ inline __bf16 f2bf(float f) {
    unsigned u = __builtin_bit_cast(unsigned, f);
    unsigned short h = (unsigned short)(u >> 16);   // truncate to bf16
    return __builtin_bit_cast(__bf16, h);
}

// 16-element bf16 fragment from two contiguous 16-byte halves
static __device__ inline bf16x16 ld_frag(const __bf16* __restrict__ p0,
                                         const __bf16* __restrict__ p1) {
    bf16x8 lo = *(const bf16x8*)p0;
    bf16x8 hi = *(const bf16x8*)p1;
    return __builtin_shufflevector(lo, hi, 0, 1, 2, 3, 4, 5, 6, 7,
                                           8, 9, 10, 11, 12, 13, 14, 15);
}

static __device__ inline unsigned long long shflx64(unsigned long long v, int m) {
    unsigned lo = (unsigned)v, hi = (unsigned)(v >> 32);
    lo = __shfl_xor(lo, m, 32);
    hi = __shfl_xor(hi, m, 32);
    return ((unsigned long long)hi << 32) | lo;
}

// ---------------------------------------------------------------------------
// f32 -> bf16 conversion kernels (run once per call, outside the GEMM loops)
// ---------------------------------------------------------------------------
__global__ void cvt_kernel(const float* __restrict__ in,
                           __bf16* __restrict__ out, long n) {
    long i = (long)blockIdx.x * 256 + threadIdx.x;
    if (i < n) out[i] = f2bf(in[i]);
}

__global__ void cvt_pad_kernel(const float* __restrict__ in,
                               __bf16* __restrict__ out,
                               int rows, int cin, int cout) {
    long i = (long)blockIdx.x * 256 + threadIdx.x;
    if (i < (long)rows * cout) {
        int r = (int)(i / cout), c = (int)(i % cout);
        out[i] = (c < cin) ? f2bf(in[(long)r * cin + c]) : f2bf(0.f);
    }
}

// ---------------------------------------------------------------------------
// init: h0 = concat(init_hidden, att_embedding) in f32 + bf16; zero argmax
// ---------------------------------------------------------------------------
__global__ void init_kernel(const float* __restrict__ init_hid,
                            const float* __restrict__ att,
                            float* __restrict__ h0f, __bf16* __restrict__ h0b,
                            unsigned long long* __restrict__ amax) {
    int idx = blockIdx.x * 256 + threadIdx.x;
    if (idx < B_ * H_) {
        int b = idx / H_, j = idx % H_;
        float v = (j < DINIT_) ? init_hid[b * DINIT_ + j]
                               : att[b * DATT_ + (j - DINIT_)];
        h0f[idx] = v;
        h0b[idx] = f2bf(v);
    }
    if (idx < B_ * T_) amax[idx] = 0ull;
}

// ---------------------------------------------------------------------------
// One GRU step. Each wave owns a 16x16 tile of [B=64, H=1024]; three WMMA
// chains (r/z/n) share the A fragment. All operands pre-converted bf16.
// ---------------------------------------------------------------------------
__global__ __launch_bounds__(256) void gru_step(
    const int*    __restrict__ inputs,
    const __bf16* __restrict__ emb_bf,    // [V][EP_]
    const __bf16* __restrict__ Wih_bf,    // [3H][EP_]
    const __bf16* __restrict__ Whh_bf,    // [3H][H]
    const float*  __restrict__ b_ih,
    const float*  __restrict__ b_hh,
    const float*  __restrict__ h_in_f,    // [B][H] f32 (for blend)
    const __bf16* __restrict__ h_in_b,    // [B][H] bf16 (A frags)
    float*        __restrict__ h_out_f,
    __bf16*       __restrict__ h_out_b,
    float*        __restrict__ hiddens,   // d_out [B][T][H]
    __bf16*       __restrict__ hidd_bf,   // ws [B*T][H]
    int t)
{
    const int wave   = (blockIdx.x * blockDim.x + threadIdx.x) >> 5;
    const int lane   = threadIdx.x & 31;
    const int laneLo = lane & 15;
    const int hi     = lane >> 4;
    const int rowTile = wave & 3;           // B/16 = 4
    const int colTile = wave >> 2;          // H/16 = 64
    const int aRow = rowTile * 16 + laneLo;
    const int nCol = colTile * 16 + laneLo;

    const int tok = inputs[aRow * (T_ + 1) + t];
    const __bf16* __restrict__ xrow = emb_bf + (long)tok * EP_;
    const __bf16* __restrict__ hrow = h_in_b + (long)aRow * H_;

    f32x8 acc_r = {}, acc_z = {}, acc_ni = {}, acc_nh = {};

    // ---- gi = x @ W_ih^T : K over EP_ (zero-padded, no masking) ----
    for (int k0 = 0; k0 < EP_; k0 += 32) {
        bf16x16 a = ld_frag(xrow + k0 + hi * 16, xrow + k0 + hi * 16 + 8);
        {
            const __bf16* w = Wih_bf + (long)(0 * H_ + nCol) * EP_ + k0 + hi * 16;
            bf16x16 bm = ld_frag(w, w + 8);
            acc_r  = __builtin_amdgcn_wmma_f32_16x16x32_bf16(false, a, false, bm, (short)0, acc_r,  false, false);
        }
        {
            const __bf16* w = Wih_bf + (long)(1 * H_ + nCol) * EP_ + k0 + hi * 16;
            bf16x16 bm = ld_frag(w, w + 8);
            acc_z  = __builtin_amdgcn_wmma_f32_16x16x32_bf16(false, a, false, bm, (short)0, acc_z,  false, false);
        }
        {
            const __bf16* w = Wih_bf + (long)(2 * H_ + nCol) * EP_ + k0 + hi * 16;
            bf16x16 bm = ld_frag(w, w + 8);
            acc_ni = __builtin_amdgcn_wmma_f32_16x16x32_bf16(false, a, false, bm, (short)0, acc_ni, false, false);
        }
    }

    // ---- gh = h @ W_hh^T : K over H ----
    for (int k0 = 0; k0 < H_; k0 += 32) {
        bf16x16 a = ld_frag(hrow + k0 + hi * 16, hrow + k0 + hi * 16 + 8);
        {
            const __bf16* w = Whh_bf + (long)(0 * H_ + nCol) * H_ + k0 + hi * 16;
            bf16x16 bm = ld_frag(w, w + 8);
            acc_r  = __builtin_amdgcn_wmma_f32_16x16x32_bf16(false, a, false, bm, (short)0, acc_r,  false, false);
        }
        {
            const __bf16* w = Whh_bf + (long)(1 * H_ + nCol) * H_ + k0 + hi * 16;
            bf16x16 bm = ld_frag(w, w + 8);
            acc_z  = __builtin_amdgcn_wmma_f32_16x16x32_bf16(false, a, false, bm, (short)0, acc_z,  false, false);
        }
        {
            const __bf16* w = Whh_bf + (long)(2 * H_ + nCol) * H_ + k0 + hi * 16;
            bf16x16 bm = ld_frag(w, w + 8);
            acc_nh = __builtin_amdgcn_wmma_f32_16x16x32_bf16(false, a, false, bm, (short)0, acc_nh, false, false);
        }
    }

    // ---- gates + blend (C/D layout: elem i -> M = 8*hi + i, N = laneLo) ----
    const float br  = b_ih[nCol]          + b_hh[nCol];
    const float bz  = b_ih[H_ + nCol]     + b_hh[H_ + nCol];
    const float bni = b_ih[2 * H_ + nCol];
    const float bnh = b_hh[2 * H_ + nCol];
    #pragma unroll
    for (int i = 0; i < 8; ++i) {
        int brow = rowTile * 16 + hi * 8 + i;
        float r = 1.f / (1.f + __expf(-(acc_r[i] + br)));
        float z = 1.f / (1.f + __expf(-(acc_z[i] + bz)));
        float n = tanhf(acc_ni[i] + bni + r * (acc_nh[i] + bnh));
        float hp = h_in_f[(long)brow * H_ + nCol];
        float hv = (1.f - z) * n + z * hp;
        h_out_f[(long)brow * H_ + nCol] = hv;
        h_out_b[(long)brow * H_ + nCol] = f2bf(hv);
        hiddens[((long)brow * T_ + t) * H_ + nCol] = hv;     // row = b*T + t
        hidd_bf[((long)brow * T_ + t) * H_ + nCol] = f2bf(hv);
    }
}

// ---------------------------------------------------------------------------
// logits = hiddens @ W_out^T + b_out, fused argmax.
// The 8 waves of a block share one colTile: the 32KB W_out panel is staged
// into LDS once per block with GLOBAL_LOAD_ASYNC_TO_LDS_B128 (ASYNCcnt),
// then B fragments come from bank-conflict-free ds_load_b128.
// ---------------------------------------------------------------------------
__global__ __launch_bounds__(256) void logits_kernel(
    const __bf16* __restrict__ hidd_bf,   // [3200][H]
    const __bf16* __restrict__ Wout_bf,   // [V][H]
    const float*  __restrict__ b_out,
    float*        __restrict__ logits,
    unsigned long long* __restrict__ amax)
{
    __shared__ __align__(16) char bpanel[16 * LDS_COL_STRIDE];  // ~32.25 KB

    const int wave   = threadIdx.x >> 5;
    const int lane   = threadIdx.x & 31;
    const int laneLo = lane & 15;
    const int hi     = lane >> 4;
    const int colTile = blockIdx.x / 25;             // 30000/16 = 1875
    const int rowTile = (blockIdx.x % 25) * 8 + wave; // 200 = 25*8
    const int aRow = rowTile * 16 + laneLo;
    const int nCol = colTile * 16 + laneLo;

    // ---- stage B panel [16 cols][1024 k] bf16 into LDS (async) ----
    {
        const __bf16* panel = Wout_bf + (size_t)colTile * 16 * H_;
        const unsigned ldsbase = (unsigned)(uintptr_t)(&bpanel[0]);
        #pragma unroll
        for (int i = 0; i < 8; ++i) {
            int flat = i * 256 + threadIdx.x;        // 0..2047 16B slots
            int c  = flat >> 7;                      // column 0..15
            int kc = flat & 127;                     // 16B chunk within column
            unsigned long long ga =
                (unsigned long long)(uintptr_t)(panel + (size_t)c * H_ + kc * 8);
            unsigned la = ldsbase + c * LDS_COL_STRIDE + kc * 16;
            asm volatile("global_load_async_to_lds_b128 %0, %1, off"
                         :: "v"(la), "v"(ga) : "memory");
        }
        asm volatile("s_wait_asynccnt 0" ::: "memory");
    }
    __syncthreads();

    const __bf16* __restrict__ arow = hidd_bf + (long)aRow * H_;
    const char*   bcol = &bpanel[laneLo * LDS_COL_STRIDE];

    f32x8 acc = {};
    for (int k0 = 0; k0 < H_; k0 += 32) {
        bf16x16 a = ld_frag(arow + k0 + hi * 16, arow + k0 + hi * 16 + 8);
        const __bf16* p = (const __bf16*)(bcol + (size_t)(k0 + hi * 16) * 2);
        bf16x16 bm = ld_frag(p, p + 8);
        if (k0 + 64 < H_) __builtin_prefetch(arow + k0 + 64, 0, 1);
        acc = __builtin_amdgcn_wmma_f32_16x16x32_bf16(false, a, false, bm, (short)0, acc, false, false);
    }

    const float bo = b_out[nCol];
    #pragma unroll
    for (int i = 0; i < 8; ++i) {
        int m = rowTile * 16 + hi * 8 + i;
        float val = acc[i] + bo;
        logits[(long)m * V_ + nCol] = val;
        // order-preserving float key; inverted index => ties pick smallest n
        unsigned u = __builtin_bit_cast(unsigned, val);
        unsigned key = (u & 0x80000000u) ? ~u : (u | 0x80000000u);
        unsigned long long packed =
            ((unsigned long long)key << 32) | (unsigned)(V_ - 1 - nCol);
        #pragma unroll
        for (int off = 8; off >= 1; off >>= 1) {
            unsigned long long o = shflx64(packed, off);
            if (o > packed) packed = o;
        }
        if (laneLo == 0) atomicMax(&amax[m], packed);
    }
}

__global__ void finalize_kernel(const unsigned long long* __restrict__ amax,
                                float* __restrict__ out_ids) {
    int i = blockIdx.x * 256 + threadIdx.x;
    if (i < B_ * T_) {
        unsigned inv = (unsigned)(amax[i] & 0xFFFFFFFFull);
        out_ids[i] = (float)(V_ - 1 - (int)inv);
    }
}

// ---------------------------------------------------------------------------
extern "C" void kernel_launch(void* const* d_in, const int* in_sizes, int n_in,
                              void* d_out, int out_size, void* d_ws, size_t ws_size,
                              hipStream_t stream)
{
    const int*   inputs   = (const int*)  d_in[0];
    // d_in[1] = max_length scalar (compile-time T_ = 50)
    const float* init_hid = (const float*)d_in[2];
    const float* att      = (const float*)d_in[3];
    const float* emb      = (const float*)d_in[4];
    const float* W_ih     = (const float*)d_in[5];
    const float* W_hh     = (const float*)d_in[6];
    const float* b_ih     = (const float*)d_in[7];
    const float* b_hh     = (const float*)d_in[8];
    const float* W_out    = (const float*)d_in[9];
    const float* b_out    = (const float*)d_in[10];

    float* out     = (float*)d_out;
    float* hiddens = out;                                  // 64*50*1024
    float* logits  = out + (size_t)B_ * T_ * H_;           // 64*50*30000
    float* sampled = logits + (size_t)B_ * T_ * V_;        // 64*50

    // ---- workspace carve (all 16B-aligned slices) ----
    char* ws = (char*)d_ws;
    __bf16* Wout_bf = (__bf16*)ws;                          ws += (size_t)V_ * H_ * 2;       // 61.44 MB
    __bf16* emb_bf  = (__bf16*)ws;                          ws += (size_t)V_ * EP_ * 2;      // 19.20 MB
    __bf16* Whh_bf  = (__bf16*)ws;                          ws += (size_t)3 * H_ * H_ * 2;   //  6.29 MB
    __bf16* Wih_bf  = (__bf16*)ws;                          ws += (size_t)3 * H_ * EP_ * 2;  //  1.97 MB
    __bf16* hidd_bf = (__bf16*)ws;                          ws += (size_t)B_ * T_ * H_ * 2;  //  6.55 MB
    float*  h0f     = (float*)ws;                           ws += (size_t)B_ * H_ * 4;
    float*  h1f     = (float*)ws;                           ws += (size_t)B_ * H_ * 4;
    __bf16* h0b     = (__bf16*)ws;                          ws += (size_t)B_ * H_ * 2;
    __bf16* h1b     = (__bf16*)ws;                          ws += (size_t)B_ * H_ * 2;
    unsigned long long* amax = (unsigned long long*)ws;

    // ---- one-time layout/precision conversions ----
    {
        long n = (long)V_ * H_;
        cvt_kernel<<<(int)((n + 255) / 256), 256, 0, stream>>>(W_out, Wout_bf, n);
        n = (long)3 * H_ * H_;
        cvt_kernel<<<(int)((n + 255) / 256), 256, 0, stream>>>(W_hh, Whh_bf, n);
        n = (long)V_ * EP_;
        cvt_pad_kernel<<<(int)((n + 255) / 256), 256, 0, stream>>>(emb, emb_bf, V_, E_, EP_);
        n = (long)3 * H_ * EP_;
        cvt_pad_kernel<<<(int)((n + 255) / 256), 256, 0, stream>>>(W_ih, Wih_bf, 3 * H_, E_, EP_);
    }

    init_kernel<<<(B_ * H_ + 255) / 256, 256, 0, stream>>>(init_hid, att, h0f, h0b, amax);

    float*  hinf = h0f, *houtf = h1f;
    __bf16* hinb = h0b; __bf16* houtb = h1b;
    for (int t = 0; t < T_; ++t) {
        gru_step<<<32, 256, 0, stream>>>(inputs, emb_bf, Wih_bf, Whh_bf,
                                         b_ih, b_hh, hinf, hinb, houtf, houtb,
                                         hiddens, hidd_bf, t);
        float* tf = hinf; hinf = houtf; houtf = tf;
        __bf16* tb = hinb; hinb = houtb; houtb = tb;
    }

    const int blocks = 25 * (V_ / 16);                     // 46875
    logits_kernel<<<blocks, 256, 0, stream>>>(hidd_bf, Wout_bf, b_out,
                                              logits, amax);

    finalize_kernel<<<(B_ * T_ + 255) / 256, 256, 0, stream>>>(amax, sampled);
}
